// AttentionModel_7473243095726
// MI455X (gfx1250) — compile-verified, matrix operand order checked by reading, not run on hardware
//
#include <hip/hip_runtime.h>

// ---------------- types ----------------
typedef __bf16 bf16_t;
typedef __attribute__((ext_vector_type(4)))  bf16_t v4bf;
typedef __attribute__((ext_vector_type(8)))  bf16_t v8bf;
typedef __attribute__((ext_vector_type(16))) bf16_t v16bf;
typedef __attribute__((ext_vector_type(8)))  float  v8f;

#define BM 128
#define BN 128
#define BK 64
#define LDSS 72   // LDS row stride in bf16 units: 64 data + 8 pad = 144B (16B aligned, conflict-free bank walk)
#define TILE_H (BM * LDSS)

// ---------------- WMMA helpers ----------------
// A-fragment layout for V_WMMA_*_16X16X32_BF16 (ISA 7.12.2, 16-bit A 16x32):
//   lanes 0-15 : row M=lane,    elems 0-7 = K 0..7,   elems 8-15 = K 16..23
//   lanes 16-31: row M=lane-16, elems 0-7 = K 8..15,  elems 8-15 = K 24..31
// B uses the symmetric layout with N in place of M (B tiles staged as [n][k] in LDS;
// all B operands are pre-transposed in HBM so staging is contiguous).
__device__ __forceinline__ v16bf load_frag(const bf16_t* base) {
  const int lane = threadIdx.x & 31;
  const int r = lane & 15;
  const int h = (lane >> 4) & 1;
  const v8bf* p = (const v8bf*)(base + r * LDSS);   // 16B-aligned: LDSS*2 = 144
  v8bf lo = p[h];          // K = 8h .. 8h+7
  v8bf hi = p[2 + h];      // K = 16+8h .. 16+8h+7
  return __builtin_shufflevector(lo, hi, 0, 1, 2, 3, 4, 5, 6, 7,
                                 8, 9, 10, 11, 12, 13, 14, 15);
}

__device__ __forceinline__ v8f wmma_bf16(v16bf a, v16bf b, v8f c) {
  // (neg_a, A, neg_b, B, c_mod, C, reuse_a, reuse_b)
  return __builtin_amdgcn_wmma_f32_16x16x32_bf16(false, a, false, b, (short)0, c,
                                                 false, false);
}

// ---------------- async LDS staging (256 threads) ----------------
// 128 rows x 64 halves, contiguous rows, via GLOBAL_LOAD_ASYNC_TO_LDS_B128:
// data goes L2 -> LDS directly (no VGPR round trip), tracked by ASYNCcnt.
// Each call issues 4 async ops per thread.
__device__ __forceinline__ void stage_tile_async(const bf16_t* __restrict__ src,
                                                 int ld, bf16_t* __restrict__ dst,
                                                 int tid) {
#pragma unroll
  for (int i = 0; i < 4; ++i) {
    int c = tid + i * 256;           // 1024 chunks of 8 halves (16B)
    int row = c >> 3;
    int seg = (c & 7) << 3;
    unsigned lds_off = (unsigned)(uintptr_t)(dst + row * LDSS + seg);
    const bf16_t* g = src + (size_t)row * ld + seg;
    asm volatile("global_load_async_to_lds_b128 %0, %1, off"
                 :: "v"(lds_off), "v"(g)
                 : "memory");
  }
}

// Async ops complete in order: after issuing the next tile pair (8 ops/thread),
// waiting to depth 8 guarantees the PREVIOUS pair has fully landed in LDS.
__device__ __forceinline__ void wait_async_prev() {
  asm volatile("s_wait_asynccnt 0x8" ::: "memory");
}
__device__ __forceinline__ void wait_async_all() {
  asm volatile("s_wait_asynccnt 0x0" ::: "memory");
}

// One BK=64 LDS tile -> 2 k-steps x 8 WMMA (wave tile 32x64).
__device__ __forceinline__ void mma_tile(const bf16_t* sA, const bf16_t* sB,
                                         int wm, int wn, v8f acc[2][4]) {
#pragma unroll
  for (int ks = 0; ks < BK; ks += 32) {
    v16bf a0 = load_frag(sA + (wm + 0) * LDSS + ks);
    v16bf a1 = load_frag(sA + (wm + 16) * LDSS + ks);
#pragma unroll
    for (int nj = 0; nj < 4; ++nj) {
      v16bf b = load_frag(sB + (wn + nj * 16) * LDSS + ks);
      acc[0][nj] = wmma_bf16(a0, b, acc[0][nj]);
      acc[1][nj] = wmma_bf16(a1, b, acc[1][nj]);
    }
  }
}

// Double-buffered async K-loop shared by all three GEMMs.
// Asrc/Bsrc point at (row0, k=0); lda/ldb are row strides in halves.
__device__ __forceinline__ void gemm_pipeline(const bf16_t* __restrict__ Asrc, int lda,
                                              const bf16_t* __restrict__ Bsrc, int ldb,
                                              int KD, bf16_t* sA, bf16_t* sB,
                                              int tid, int wm, int wn, v8f acc[2][4]) {
  // prologue: group 0 -> buffer 0
  stage_tile_async(Asrc, lda, sA, tid);
  stage_tile_async(Bsrc, ldb, sB, tid);
  int cur = 0;
  for (int k0 = 0; k0 < KD; k0 += BK, cur ^= 1) {
    if (k0 + BK < KD) {
      // issue group i+1 into the other buffer, then wait for group i only
      stage_tile_async(Asrc + k0 + BK, lda, sA + (cur ^ 1) * TILE_H, tid);
      stage_tile_async(Bsrc + k0 + BK, ldb, sB + (cur ^ 1) * TILE_H, tid);
      wait_async_prev();
    } else {
      wait_async_all();
    }
    __syncthreads();                       // all waves' current tile landed
    mma_tile(sA + cur * TILE_H, sB + cur * TILE_H, wm, wn, acc);
    __syncthreads();                       // done reading before next overwrite
  }
}

// ---------------- fp32 -> bf16 convert (row-major) ----------------
__global__ __launch_bounds__(256) void k_cvt(const float* __restrict__ in,
                                             bf16_t* __restrict__ out, int n4) {
  int i = blockIdx.x * blockDim.x + threadIdx.x;
  int stride = gridDim.x * blockDim.x;
  for (; i < n4; i += stride) {
    float4 f = ((const float4*)in)[i];
    v4bf o;
    o[0] = (bf16_t)f.x; o[1] = (bf16_t)f.y; o[2] = (bf16_t)f.z; o[3] = (bf16_t)f.w;
    ((v4bf*)out)[i] = o;
  }
}

// ---------------- fp32 [1024][1024] -> bf16 transposed [n][k] ----------------
__global__ __launch_bounds__(256) void k_cvtT(const float* __restrict__ in,
                                              bf16_t* __restrict__ out) {
  __shared__ float t[32][33];
  const int c0 = blockIdx.x * 32, r0 = blockIdx.y * 32;
  const int tx = threadIdx.x & 31, ty = threadIdx.x >> 5;   // 32 x 8
#pragma unroll
  for (int j = ty; j < 32; j += 8)
    t[j][tx] = in[(size_t)(r0 + j) * 1024 + c0 + tx];
  __syncthreads();
#pragma unroll
  for (int j = ty; j < 32; j += 8)
    out[(size_t)(c0 + j) * 1024 + r0 + tx] = (bf16_t)t[tx][j];
}

// ---------------- bf16 V [t][d] -> Vt [d][t] per batch ----------------
__global__ __launch_bounds__(256) void k_transV(const bf16_t* __restrict__ in,
                                                bf16_t* __restrict__ out) {
  __shared__ bf16_t t[32][34];  // +2 halves pad
  const int b = blockIdx.z;
  const bf16_t* src = in + (size_t)b * 2048 * 1024;
  bf16_t* dst = out + (size_t)b * 2048 * 1024;
  const int c0 = blockIdx.x * 32, r0 = blockIdx.y * 32;
  const int tx = threadIdx.x & 31, ty = threadIdx.x >> 5;
#pragma unroll
  for (int j = ty; j < 32; j += 8)
    t[j][tx] = src[(size_t)(r0 + j) * 1024 + c0 + tx];
  __syncthreads();
#pragma unroll
  for (int j = ty; j < 32; j += 8)
    dst[(size_t)(c0 + j) * 2048 + r0 + tx] = t[tx][j];
}

// ---------------- QKV projection GEMM: O = X @ W + b (W pre-transposed) ----------------
__global__ __launch_bounds__(256) void k_gemm_qkv(
    const bf16_t* __restrict__ Xb, const bf16_t* __restrict__ Wt,
    const float* __restrict__ bq, const float* __restrict__ bk,
    const float* __restrict__ bv,
    bf16_t* __restrict__ Qb, bf16_t* __restrict__ Kb, bf16_t* __restrict__ Vb,
    float* __restrict__ Vf) {
  __shared__ bf16_t sA[2 * TILE_H];
  __shared__ bf16_t sB[2 * TILE_H];
  const int tid = threadIdx.x, lane = tid & 31, wid = tid >> 5;
  const int wm = (wid >> 1) * 32, wn = (wid & 1) * 64;
  const int m0 = blockIdx.y * BM, n0 = blockIdx.x * BN, z = blockIdx.z;
  const bf16_t* W = Wt + (size_t)z * 1024 * 1024;   // [n][k]
  const float* bias = (z == 0) ? bq : ((z == 1) ? bk : bv);
  bf16_t* Ob = (z == 0) ? Qb : ((z == 1) ? Kb : Vb);

  v8f acc[2][4] = {};
  gemm_pipeline(Xb + (size_t)m0 * 1024, 1024,
                W + (size_t)n0 * 1024, 1024, 1024,
                sA, sB, tid, wm, wn, acc);

  const int rb = (lane >> 4) << 3, cc = lane & 15;
#pragma unroll
  for (int mi = 0; mi < 2; ++mi)
#pragma unroll
    for (int nj = 0; nj < 4; ++nj) {
      int col = n0 + wn + nj * 16 + cc;
      float bcol = bias[col];
#pragma unroll
      for (int i = 0; i < 8; ++i) {
        int row = m0 + wm + mi * 16 + rb + i;
        float val = acc[mi][nj][i] + bcol;
        Ob[(size_t)row * 1024 + col] = (bf16_t)val;
        if (z == 2) Vf[(size_t)row * 1024 + col] = val;
      }
    }
}

// ---------------- scores = Q K^T / 32, masked ----------------
__global__ __launch_bounds__(256) void k_scores(
    const bf16_t* __restrict__ Qb, const bf16_t* __restrict__ Kb,
    const long long* __restrict__ seqlen, float* __restrict__ S) {
  __shared__ bf16_t sA[2 * TILE_H];
  __shared__ bf16_t sB[2 * TILE_H];
  const int tid = threadIdx.x, lane = tid & 31, wid = tid >> 5;
  const int wm = (wid >> 1) * 32, wn = (wid & 1) * 64;
  const int n0 = blockIdx.x * BN, m0 = blockIdx.y * BM, b = blockIdx.z;
  const bf16_t* Q = Qb + (size_t)b * 2048 * 1024;
  const bf16_t* K = Kb + (size_t)b * 2048 * 1024;

  v8f acc[2][4] = {};
  gemm_pipeline(Q + (size_t)m0 * 1024, 1024,
                K + (size_t)n0 * 1024, 1024, 1024,   // K rows ARE B columns
                sA, sB, tid, wm, wn, acc);

  const long long len = seqlen[b];
  float* Sb = S + (size_t)b * 2048 * 2048;
  const int rb = (lane >> 4) << 3, cc = lane & 15;
#pragma unroll
  for (int mi = 0; mi < 2; ++mi)
#pragma unroll
    for (int nj = 0; nj < 4; ++nj) {
      int t = n0 + wn + nj * 16 + cc;
      bool ok = (long long)t < len;
#pragma unroll
      for (int i = 0; i < 8; ++i) {
        int m = m0 + wm + mi * 16 + rb + i;
        float v = ok ? acc[mi][nj][i] * 0.03125f : -3.0e38f;
        Sb[(size_t)m * 2048 + t] = v;
      }
    }
}

// ---------------- row softmax, fp32 scores -> bf16 probs in place ----------------
__global__ __launch_bounds__(256) void k_softmax(const float* __restrict__ S,
                                                 bf16_t* __restrict__ P) {
  const int tid = threadIdx.x, lane = tid & 31, wid = tid >> 5;
  const size_t row = (size_t)blockIdx.y * 2048 + blockIdx.x;
  const float* src = S + row * 2048;
  bf16_t* dst = P + row * 4096;  // same byte offset as fp32 row start
  __shared__ float redm[8], reds[8];

  float v[8], m = -3.4e38f;
#pragma unroll
  for (int i = 0; i < 8; ++i) { v[i] = src[tid + (i << 8)]; m = fmaxf(m, v[i]); }
#pragma unroll
  for (int off = 16; off > 0; off >>= 1) m = fmaxf(m, __shfl_xor(m, off, 32));
  if (lane == 0) redm[wid] = m;
  __syncthreads();
  m = redm[0];
#pragma unroll
  for (int j = 1; j < 8; ++j) m = fmaxf(m, redm[j]);

  float s = 0.f;
#pragma unroll
  for (int i = 0; i < 8; ++i) { v[i] = __expf(v[i] - m); s += v[i]; }
#pragma unroll
  for (int off = 16; off > 0; off >>= 1) s += __shfl_xor(s, off, 32);
  if (lane == 0) reds[wid] = s;
  __syncthreads();
  s = 0.f;
#pragma unroll
  for (int j = 0; j < 8; ++j) s += reds[j];
  const float inv = 1.0f / s;
#pragma unroll
  for (int i = 0; i < 8; ++i) dst[tid + (i << 8)] = (bf16_t)(v[i] * inv);
}

// ---------------- out = P @ V + V (V pre-transposed to [d][t]) ----------------
__global__ __launch_bounds__(256) void k_out(
    const bf16_t* __restrict__ P, const bf16_t* __restrict__ Vt,
    const float* __restrict__ Vf, float* __restrict__ out) {
  __shared__ bf16_t sA[2 * TILE_H];
  __shared__ bf16_t sB[2 * TILE_H];
  const int tid = threadIdx.x, lane = tid & 31, wid = tid >> 5;
  const int wm = (wid >> 1) * 32, wn = (wid & 1) * 64;
  const int n0 = blockIdx.x * BN, m0 = blockIdx.y * BM, b = blockIdx.z;
  const bf16_t* Pb = P + (size_t)b * 2048 * 4096;    // bf16 probs, row stride 4096
  const bf16_t* V = Vt + (size_t)b * 2048 * 1024;    // [d=1024][t=2048]

  v8f acc[2][4] = {};
  gemm_pipeline(Pb + (size_t)m0 * 4096, 4096,
                V + (size_t)n0 * 2048, 2048, 2048,
                sA, sB, tid, wm, wn, acc);

  const int rb = (lane >> 4) << 3, cc = lane & 15;
#pragma unroll
  for (int mi = 0; mi < 2; ++mi)
#pragma unroll
    for (int nj = 0; nj < 4; ++nj) {
      int col = n0 + wn + nj * 16 + cc;
#pragma unroll
      for (int i = 0; i < 8; ++i) {
        int row = m0 + wm + mi * 16 + rb + i;
        size_t idx = ((size_t)b * 2048 + row) * 1024 + col;
        out[idx] = acc[mi][nj][i] + Vf[idx];
      }
    }
}

// ---------------- host launcher ----------------
extern "C" void kernel_launch(void* const* d_in, const int* in_sizes, int n_in,
                              void* d_out, int out_size, void* d_ws, size_t ws_size,
                              hipStream_t stream) {
  const float* plms1 = (const float*)d_in[0];
  const float* Wq = (const float*)d_in[1];
  const float* bq = (const float*)d_in[2];
  const float* Wk = (const float*)d_in[3];
  const float* bk = (const float*)d_in[4];
  const float* Wv = (const float*)d_in[5];
  const float* bv = (const float*)d_in[6];
  const long long* seqlen = (const long long*)d_in[7];
  float* out = (float*)d_out;

  // workspace carve-up (256B aligned slabs)
  char* ws = (char*)d_ws;
  size_t off = 0;
  auto take = [&](size_t bytes) -> char* {
    char* p = ws + off;
    off += (bytes + 255) & ~(size_t)255;
    return p;
  };
  const size_t MD = (size_t)16384 * 1024;   // B*S x D
  bf16_t* Xb = (bf16_t*)take(MD * 2);                      // 32 MB
  bf16_t* Wt = (bf16_t*)take((size_t)3 * 1024 * 1024 * 2); // 6 MB (transposed)
  bf16_t* Qb = (bf16_t*)take(MD * 2);                      // 32 MB
  bf16_t* Kb = (bf16_t*)take(MD * 2);                      // 32 MB
  bf16_t* Vb = (bf16_t*)take(MD * 2);                      // 32 MB  [t][d]
  bf16_t* Vtb = (bf16_t*)take(MD * 2);                     // 32 MB  [d][t]
  float*  Vf = (float*)take(MD * 4);                       // 64 MB
  float*  S  = (float*)take((size_t)8 * 2048 * 2048 * 4);  // 128 MB
  bf16_t* P  = (bf16_t*)S;  // bf16 probs written in place over fp32 scores

  // 1) fp32 -> bf16 conversion (X row-major; W transposed to [n][k])
  k_cvt<<<4096, 256, 0, stream>>>(plms1, Xb, (int)(MD / 4));
  k_cvtT<<<dim3(32, 32), 256, 0, stream>>>(Wq, Wt + (size_t)0 * 1024 * 1024);
  k_cvtT<<<dim3(32, 32), 256, 0, stream>>>(Wk, Wt + (size_t)1 * 1024 * 1024);
  k_cvtT<<<dim3(32, 32), 256, 0, stream>>>(Wv, Wt + (size_t)2 * 1024 * 1024);

  // 2) Q/K/V projections (z selects which)
  k_gemm_qkv<<<dim3(8, 128, 3), 256, 0, stream>>>(Xb, Wt, bq, bk, bv, Qb, Kb, Vb, Vf);

  // 3) V -> Vt [d][t] for contiguous B staging in the PV GEMM
  k_transV<<<dim3(32, 64, 8), 256, 0, stream>>>(Vb, Vtb);

  // 4) masked scores
  k_scores<<<dim3(16, 16, 8), 256, 0, stream>>>(Qb, Kb, seqlen, S);

  // 5) softmax rows (fp32 in, bf16 out in place)
  k_softmax<<<dim3(2048, 8), 256, 0, stream>>>(S, P);

  // 6) out = P @ V + V
  k_out<<<dim3(8, 16, 8), 256, 0, stream>>>(P, Vtb, Vf, out);

  (void)in_sizes; (void)n_in; (void)out_size; (void)ws_size;
}
